// SelfAttention_65008624992334
// MI455X (gfx1250) — compile-verified, matrix operand order checked by reading, not run on hardware
//
#include <hip/hip_runtime.h>
#include <hip/hip_bf16.h>
#include <stdint.h>

typedef __attribute__((ext_vector_type(16))) __bf16 v16bf;
typedef __attribute__((ext_vector_type(8)))  float  v8f;

#define B_  4
#define H_  16
#define S_  1024
#define D_  256
#define A_  128
#define O_  256
#define BH_ (B_*H_)

__device__ __forceinline__ unsigned short f2bf(float f) {
  unsigned int u = __float_as_uint(f);
  u += 0x7FFFu + ((u >> 16) & 1u);            // round-to-nearest-even
  return (unsigned short)(u >> 16);
}

__device__ __forceinline__ v16bf ldfrag(const unsigned short* p) {
  return *reinterpret_cast<const v16bf*>(p);  // 32B contiguous fragment
}
__device__ __forceinline__ v8f wmma_bf16(v16bf a, v16bf b, v8f c) {
  return __builtin_amdgcn_wmma_f32_16x16x32_bf16(false, a, false, b, (short)0, c, false, false);
}

// ---------------- LayerNorm: one wave per row of 256, bf16 output --------
__global__ void ln_kernel(const float* __restrict__ x,
                          const float* __restrict__ gamma,
                          const float* __restrict__ beta,
                          unsigned short* __restrict__ xn) {
  const int w = threadIdx.x >> 5, lane = threadIdx.x & 31;
  const long row = (long)blockIdx.x * 8 + w;
  const float* xr = x + row * D_ + lane * 8;
  float4 a = *(const float4*)(xr);
  float4 b = *(const float4*)(xr + 4);
  float s  = a.x + a.y + a.z + a.w + b.x + b.y + b.z + b.w;
  float ss = a.x*a.x + a.y*a.y + a.z*a.z + a.w*a.w
           + b.x*b.x + b.y*b.y + b.z*b.z + b.w*b.w;
#pragma unroll
  for (int off = 16; off >= 1; off >>= 1) {
    s  += __shfl_xor(s,  off, 32);
    ss += __shfl_xor(ss, off, 32);
  }
  const float mean = s * (1.0f / D_);
  const float var  = ss * (1.0f / D_) - mean * mean;
  const float rstd = rsqrtf(var + 1e-5f);
  float4 g0 = *(const float4*)(gamma + lane * 8);
  float4 g1 = *(const float4*)(gamma + lane * 8 + 4);
  float4 b0 = *(const float4*)(beta  + lane * 8);
  float4 b1 = *(const float4*)(beta  + lane * 8 + 4);
  float o[8];
  o[0] = (a.x - mean) * rstd * g0.x + b0.x;
  o[1] = (a.y - mean) * rstd * g0.y + b0.y;
  o[2] = (a.z - mean) * rstd * g0.z + b0.z;
  o[3] = (a.w - mean) * rstd * g0.w + b0.w;
  o[4] = (b.x - mean) * rstd * g1.x + b1.x;
  o[5] = (b.y - mean) * rstd * g1.y + b1.y;
  o[6] = (b.z - mean) * rstd * g1.z + b1.z;
  o[7] = (b.w - mean) * rstd * g1.w + b1.w;
  uint4 pk;
  pk.x = (unsigned)f2bf(o[0]) | ((unsigned)f2bf(o[1]) << 16);
  pk.y = (unsigned)f2bf(o[2]) | ((unsigned)f2bf(o[3]) << 16);
  pk.z = (unsigned)f2bf(o[4]) | ((unsigned)f2bf(o[5]) << 16);
  pk.w = (unsigned)f2bf(o[6]) | ((unsigned)f2bf(o[7]) << 16);
  *(uint4*)(xn + row * D_ + lane * 8) = pk;
}

// ---------------- fp32 -> bf16 weight conversion -------------------------
__global__ void cvt_kernel(const float* __restrict__ in, unsigned short* __restrict__ out, int n) {
  int i = blockIdx.x * blockDim.x + threadIdx.x;
  if (i < n) out[i] = f2bf(in[i]);
}

// ---------------- QKV projection GEMM: y[s,n] = sum_d xn[s,d]*W[n,d] -----
// block = 8 waves -> 64 (M) x 128 (N) tile; per wave 16x64 via 4 WMMA accums
// TRANSV: emit output transposed as vT[bh][n][s] via an LDS transpose tile
template <bool TRANSV>
__global__ void __launch_bounds__(256) qkv_gemm_kernel(
    const unsigned short* __restrict__ xn,
    const unsigned short* __restrict__ W,
    unsigned short* __restrict__ y, int N) {
  const int tid = threadIdx.x, w = tid >> 5, lane = tid & 31;
  const int lane15 = lane & 15, half = lane >> 4;
  const int bh = blockIdx.z, h = bh & (H_ - 1);
  const int m0b = blockIdx.x * 64, n0b = blockIdx.y * 128;
  const int m0 = m0b + (w >> 1) * 16;
  const int n0 = n0b + (w & 1) * 64;
  const unsigned short* Wh   = W + (long)h * N * D_;
  const unsigned short* Arow = xn + ((long)(bh * S_ + m0 + lane15)) * D_ + half * 16;
  const v8f vz = {0.f, 0.f, 0.f, 0.f, 0.f, 0.f, 0.f, 0.f};
  v8f acc[4];
#pragma unroll
  for (int i = 0; i < 4; ++i) acc[i] = vz;
#pragma unroll 2
  for (int k0 = 0; k0 < D_; k0 += 32) {
    v16bf a = ldfrag(Arow + k0);
    __builtin_prefetch(Arow + k0 + 32, 0, 1);   // global_prefetch
#pragma unroll
    for (int nt = 0; nt < 4; ++nt) {
      v16bf b = ldfrag(Wh + (long)(n0 + nt * 16 + lane15) * D_ + k0 + half * 16);
      acc[nt] = wmma_bf16(a, b, acc[nt]);
    }
  }
  if constexpr (!TRANSV) {
#pragma unroll
    for (int nt = 0; nt < 4; ++nt)
#pragma unroll
      for (int j = 0; j < 8; ++j) {
        int row = m0 + j + 8 * half;
        y[((long)(bh * S_ + row)) * N + n0 + nt * 16 + lane15] = f2bf(acc[nt][j]);
      }
  } else {
    // transpose 64x128 C tile through LDS, write vT[bh][n][s] coalesced
    __shared__ alignas(64) unsigned short Ct[64][136];   // +8 pad vs bank conflicts
    const int lr0 = (w >> 1) * 16, lc0 = (w & 1) * 64;
#pragma unroll
    for (int nt = 0; nt < 4; ++nt)
#pragma unroll
      for (int j = 0; j < 8; ++j)
        Ct[lr0 + j + 8 * half][lc0 + nt * 16 + lane15] = f2bf(acc[nt][j]);
    __syncthreads();
    const int lo = tid >> 1;                 // local n: 0..127
    const int sc = (tid & 1) * 32;           // local s chunk
    unsigned short* dst = y + ((long)(bh * N + n0b + lo)) * S_ + m0b + sc;
#pragma unroll
    for (int c = 0; c < 4; ++c) {
      unsigned short e[8];
#pragma unroll
      for (int i = 0; i < 8; ++i) e[i] = Ct[sc + c * 8 + i][lo];
      uint4 p;
      p.x = (unsigned)e[0] | ((unsigned)e[1] << 16);
      p.y = (unsigned)e[2] | ((unsigned)e[3] << 16);
      p.z = (unsigned)e[4] | ((unsigned)e[5] << 16);
      p.w = (unsigned)e[6] | ((unsigned)e[7] << 16);
      *(uint4*)(dst + c * 8) = p;
    }
  }
}

// ---------------- Attention: per (bh, 32-row block), two passes ----------
// s_norm is scale-invariant -> the 1/sqrt(D) factor cancels and is dropped.
__global__ void __launch_bounds__(256, 1) attn_kernel(
    const unsigned short* __restrict__ q,
    const unsigned short* __restrict__ k,
    const unsigned short* __restrict__ vT,   // [bh][O][S]
    float* __restrict__ out) {
  __shared__ alignas(64) unsigned short qs[32 * A_];   // 8 KB: q block
  __shared__ alignas(64) unsigned short Pe[32 * 64];   // exp-weights tile (bf16)
  __shared__ alignas(64) unsigned short Ps[32 * 64];   // sigmoid-weights tile (bf16)
  __shared__ float partMax[32][4];
  __shared__ float partMin[32][4];
  __shared__ float partSum[32][4];
  __shared__ float rowMax[32], rowMin[32], rowSum[32];

  const int tid = threadIdx.x, w = tid >> 5, lane = tid & 31;
  const int lane15 = lane & 15, half = lane >> 4;
  const int mi = w >> 2, ni = w & 3;                   // 2x4 wave grid
  const int bh = blockIdx.y;
  const int r0 = blockIdx.x * 32;
  const unsigned short* qh  = q  + (long)bh * S_ * A_;
  const unsigned short* kh  = k  + (long)bh * S_ * A_;
  const unsigned short* vTh = vT + (long)bh * O_ * S_;
  const v8f vz = {0.f, 0.f, 0.f, 0.f, 0.f, 0.f, 0.f, 0.f};

  { // stage q block [32 x 128] into LDS via async DMA (CDNA5 async-to-LDS path)
    const unsigned ldso = (unsigned)(uintptr_t)(&qs[0]);
    const unsigned long long gbase = (unsigned long long)(uintptr_t)(qh + (long)r0 * A_);
#pragma unroll
    for (int i = 0; i < 2; ++i) {
      unsigned           l  = ldso  + (unsigned)(tid + i * 256) * 16u;
      unsigned long long ga = gbase + (unsigned long long)(tid + i * 256) * 16ull;
      asm volatile("global_load_async_to_lds_b128 %0, %1, off" :: "v"(l), "v"(ga) : "memory");
    }
    asm volatile("s_wait_asynccnt 0" ::: "memory");
  }
  __syncthreads();

  const unsigned short* Ap = qs + (mi * 16 + lane15) * A_ + half * 16;

  // ---- pass 1: per-row max/min of raw scores (lane-wise running stats,
  //      single cross-lane reduction after the loop) ----
  {
    v8f rmaxv, rminv;
#pragma unroll
    for (int j = 0; j < 8; ++j) { rmaxv[j] = -3.0e38f; rminv[j] = 3.0e38f; }
    for (int t = 0; t < S_; t += 64) {
      v8f acc = vz;
      const unsigned short* Bp = kh + (long)(t + ni * 16 + lane15) * A_ + half * 16;
#pragma unroll
      for (int k0 = 0; k0 < A_; k0 += 32)
        acc = wmma_bf16(ldfrag(Ap + k0), ldfrag(Bp + k0), acc);
#pragma unroll
      for (int j = 0; j < 8; ++j) {
        rmaxv[j] = fmaxf(rmaxv[j], acc[j]);
        rminv[j] = fminf(rminv[j], acc[j]);
      }
    }
#pragma unroll
    for (int j = 0; j < 8; ++j) {
      float mx = rmaxv[j], mn = rminv[j];
#pragma unroll
      for (int off = 8; off >= 1; off >>= 1) {
        mx = fmaxf(mx, __shfl_xor(mx, off, 16));
        mn = fminf(mn, __shfl_xor(mn, off, 16));
      }
      if (lane15 == 0) {
        partMax[mi * 16 + j + 8 * half][ni] = mx;
        partMin[mi * 16 + j + 8 * half][ni] = mn;
      }
    }
  }
  __syncthreads();
  if (tid < 32) {
    float mx = partMax[tid][0], mn = partMin[tid][0];
#pragma unroll
    for (int i = 1; i < 4; ++i) {
      mx = fmaxf(mx, partMax[tid][i]);
      mn = fminf(mn, partMin[tid][i]);
    }
    rowMax[tid] = mx; rowMin[tid] = mn;
  }
  __syncthreads();

  // ---- pass 2: recompute scores, blend-weight, accumulate exp/sig V-products
  // per-row affine constants hoisted: sn = acc*a1 + a0
  float a0c[8], a1c[8];
#pragma unroll
  for (int j = 0; j < 8; ++j) {
    int row = mi * 16 + j + 8 * half;
    float mn = rowMin[row];
    float inv = 1.0f / (rowMax[row] - mn);
    a1c[j] = inv;
    a0c[j] = -mn * inv;
  }

  v8f accE[4], accS[4];
#pragma unroll
  for (int i = 0; i < 4; ++i) { accE[i] = vz; accS[i] = vz; }
  float sums[8];
#pragma unroll
  for (int j = 0; j < 8; ++j) sums[j] = 0.f;

  for (int t = 0; t < S_; t += 64) {
    { // phase A: all 8 waves compute a 16x16 score tile, produce P tiles
      v8f acc = vz;
      const unsigned short* Bp = kh + (long)(t + ni * 16 + lane15) * A_ + half * 16;
#pragma unroll
      for (int k0 = 0; k0 < A_; k0 += 32)
        acc = wmma_bf16(ldfrag(Ap + k0), ldfrag(Bp + k0), acc);
#pragma unroll
      for (int j = 0; j < 8; ++j) {
        int row = mi * 16 + j + 8 * half;
        float sn = fmaf(acc[j], a1c[j], a0c[j]);
        float pe = __expf(sn);
        float sg = 1.0f / (1.0f + __expf(5.0f - 10.0f * sn));  // sigmoid(10*sn-5)
        sums[j] += pe;                                         // lane-wise partial
        Pe[row * 64 + ni * 16 + lane15] = f2bf(pe);
        Ps[row * 64 + ni * 16 + lane15] = f2bf(sg);
      }
    }
    __syncthreads();
    { // phase B: P x V, B-fragments straight from L2-resident vT (contiguous)
#pragma unroll
      for (int kh2 = 0; kh2 < 2; ++kh2) {
        v16bf ae = ldfrag(Pe + (mi * 16 + lane15) * 64 + kh2 * 32 + half * 16);
        v16bf as = ldfrag(Ps + (mi * 16 + lane15) * 64 + kh2 * 32 + half * 16);
#pragma unroll
        for (int nt = 0; nt < 4; ++nt) {
          const unsigned short* bp =
              vTh + (long)(ni * 64 + nt * 16 + lane15) * S_ + t + kh2 * 32 + half * 16;
          v16bf bv = ldfrag(bp);
          accE[nt] = wmma_bf16(ae, bv, accE[nt]);
          accS[nt] = wmma_bf16(as, bv, accS[nt]);
        }
      }
    }
    __syncthreads();
  }

  // single cross-lane reduction of the exp row-sums
#pragma unroll
  for (int j = 0; j < 8; ++j) {
    float ps = sums[j];
#pragma unroll
    for (int off = 8; off >= 1; off >>= 1) ps += __shfl_xor(ps, off, 16);
    if (lane15 == 0) partSum[mi * 16 + j + 8 * half][ni] = ps;
  }
  __syncthreads();
  if (tid < 32)
    rowSum[tid] = partSum[tid][0] + partSum[tid][1] + partSum[tid][2] + partSum[tid][3];
  __syncthreads();

  const float sig5 = 1.0f / (1.0f + __expf(-5.0f));
  const float selSoft = 1.0f - sig5;                   // max(s_norm) == 1 identically
  const float selSig  = sig5;
  float* outh = out + (long)bh * S_ * O_;
#pragma unroll
  for (int nt = 0; nt < 4; ++nt)
#pragma unroll
    for (int j = 0; j < 8; ++j) {
      int row = mi * 16 + j + 8 * half;
      float val = selSoft * accE[nt][j] / rowSum[row] + selSig * accS[nt][j];
      outh[(long)(r0 + row) * O_ + ni * 64 + nt * 16 + lane15] = val;
    }
}

extern "C" void kernel_launch(void* const* d_in, const int* in_sizes, int n_in,
                              void* d_out, int out_size, void* d_ws, size_t ws_size,
                              hipStream_t stream) {
  const float* x     = (const float*)d_in[0];
  const float* Wq    = (const float*)d_in[1];
  const float* Wk    = (const float*)d_in[2];
  const float* Wv    = (const float*)d_in[3];
  const float* gamma = (const float*)d_in[4];
  const float* beta  = (const float*)d_in[5];
  float* out = (float*)d_out;

  char* ws = (char*)d_ws;
  size_t off = 0;
  auto alloc = [&](size_t bytes) -> void* {
    void* p = ws + off;
    off += (bytes + 255) & ~(size_t)255;
    return p;
  };
  unsigned short* xn = (unsigned short*)alloc((size_t)BH_ * S_ * D_ * 2);  // 32 MB
  unsigned short* wq = (unsigned short*)alloc((size_t)H_ * A_ * D_ * 2);
  unsigned short* wk = (unsigned short*)alloc((size_t)H_ * A_ * D_ * 2);
  unsigned short* wv = (unsigned short*)alloc((size_t)H_ * O_ * D_ * 2);
  unsigned short* qb = (unsigned short*)alloc((size_t)BH_ * S_ * A_ * 2);  // 16 MB
  unsigned short* kb = (unsigned short*)alloc((size_t)BH_ * S_ * A_ * 2);  // 16 MB
  unsigned short* vb = (unsigned short*)alloc((size_t)BH_ * O_ * S_ * 2);  // 32 MB (vT)

  ln_kernel<<<BH_ * S_ / 8, 256, 0, stream>>>(x, gamma, beta, xn);
  const int nqk = H_ * A_ * D_;
  const int nv  = H_ * O_ * D_;
  cvt_kernel<<<(nqk + 255) / 256, 256, 0, stream>>>(Wq, wq, nqk);
  cvt_kernel<<<(nqk + 255) / 256, 256, 0, stream>>>(Wk, wk, nqk);
  cvt_kernel<<<(nv + 255) / 256, 256, 0, stream>>>(Wv, wv, nv);
  qkv_gemm_kernel<false><<<dim3(S_ / 64, A_ / 128, BH_), 256, 0, stream>>>(xn, wq, qb, A_);
  qkv_gemm_kernel<false><<<dim3(S_ / 64, A_ / 128, BH_), 256, 0, stream>>>(xn, wk, kb, A_);
  qkv_gemm_kernel<true ><<<dim3(S_ / 64, O_ / 128, BH_), 256, 0, stream>>>(xn, wv, vb, O_);
  attn_kernel<<<dim3(S_ / 32, BH_), 256, 0, stream>>>(qb, kb, vb, out);
}
